// T5Attention_88184268521511
// MI455X (gfx1250) — matrix operand order repaired and checked by reading, not compiled, not run on hardware
//
#include <hip/hip_runtime.h>
#include <hip/hip_bf16.h>
#include <math.h>

// ---------------- problem constants ----------------
#define Bb 2
#define Ss 2048
#define Dd 1024
#define Hh 16
#define DKk 64
#define Mm (Bb * Ss)          // 4096 rows
#define LUT_LEN (2 * Ss - 1)  // 4095
#define LOG2E 1.4426950408889634f

typedef __bf16 bf16_t;
typedef __attribute__((ext_vector_type(16))) __bf16 v16bf;
typedef __attribute__((ext_vector_type(8)))  __bf16 v8bf;
typedef __attribute__((ext_vector_type(2)))  __bf16 v2bf;
typedef __attribute__((ext_vector_type(8)))  float  v8f;

// ---------------- helpers ----------------
__device__ inline bf16_t f2bf(float f) {
    unsigned u = __builtin_bit_cast(unsigned, f);
    unsigned r = u + 0x7FFFu + ((u >> 16) & 1u);     // round-to-nearest-even
    unsigned short h = (unsigned short)(r >> 16);
    return __builtin_bit_cast(bf16_t, h);
}

__device__ inline v16bf ld_pair(const bf16_t* p0, const bf16_t* p1) {
    v8bf lo = *(const v8bf*)p0;
    v8bf hi = *(const v8bf*)p1;
    v16bf r;
#pragma unroll
    for (int i = 0; i < 8; ++i) { r[i] = lo[i]; r[i + 8] = hi[i]; }
    return r;
}
__device__ inline v16bf ld16(const bf16_t* p) { return ld_pair(p, p + 8); }

__device__ inline v8f wmma_bf16(v16bf a, v16bf b, v8f c) {
    return __builtin_amdgcn_wmma_f32_16x16x32_bf16(
        /*neg_a=*/false, a, /*neg_b=*/false, b,
        /*c_mod=*/(short)0, c, /*reuse_a=*/false, /*reuse_b=*/false);
}

// max-reduction within 16-lane half-waves (rows r / r+8 live in different halves)
__device__ inline float rmax16(float v) {
#pragma unroll
    for (int off = 8; off; off >>= 1) v = fmaxf(v, __shfl_xor(v, off, 16));
    return v;
}

// ---------------- kernel 1: fp32 -> bf16 ----------------
__global__ void cvt_f32_bf16(const float* __restrict__ src, bf16_t* __restrict__ dst, int n) {
    int i = blockIdx.x * blockDim.x + threadIdx.x;
    if (i < n) dst[i] = f2bf(src[i]);
}

// ---------------- kernel 2: T5 relative-bias LUT (pre-scaled by log2e) ----------------
// lut[h][d], d = (k - q) + (S-1), faithful to _relative_position_bucket
__global__ void bias_lut_kernel(const float* __restrict__ rel_emb, float* __restrict__ lut) {
    int idx = blockIdx.x * blockDim.x + threadIdx.x;
    if (idx >= Hh * LUT_LEN) return;
    int h   = idx / LUT_LEN;
    int d   = idx % LUT_LEN;
    int rel = d - (Ss - 1);   // k - q
    int n   = -rel;
    int ret = (n < 0) ? 16 : 0;
    n = (n < 0) ? -n : n;
    int bucket;
    if (n < 8) {
        bucket = ret + n;
    } else {
        float v = logf((float)n * (1.0f / 8.0f)) * (8.0f / logf(16.0f));
        int vil = 8 + (int)v;         // trunc toward zero, matches astype(int32)
        if (vil > 15) vil = 15;
        bucket = ret + vil;
    }
    lut[idx] = rel_emb[bucket * Hh + h] * LOG2E;   // base-2 softmax pipeline
}

// ---------------- kernel 3: y = alpha * (A @ W^T), WMMA bf16 ----------------
// A: [M,K] bf16 row-major, W: [N,K] bf16 row-major (nn.Linear convention).
// One wave per 16x64 output tile; 4 f32 accumulators, K-loop step 32.
template <bool OUT_F32>
__global__ __launch_bounds__(128)
void linear_wmma(const bf16_t* __restrict__ A, const bf16_t* __restrict__ W,
                 void* __restrict__ out, float alpha, int M, int N, int K) {
    const int lane   = threadIdx.x & 31;
    const int wave   = threadIdx.x >> 5;
    const int waveId = blockIdx.x * 4 + wave;
    const int nTiles = N / 64;
    const int m0 = (waveId / nTiles) * 16;
    const int n0 = (waveId % nTiles) * 64;
    const int half = lane >> 4;        // 0: lanes 0-15, 1: lanes 16-31
    const int l16  = lane & 15;

    const bf16_t* aRow = A + (size_t)(m0 + l16) * K + half * 16;
    v8f acc[4] = {};
    for (int k0 = 0; k0 < K; k0 += 32) {
        v16bf a = ld16(aRow + k0);   // A 16x32 fragment
#pragma unroll
        for (int nc = 0; nc < 4; ++nc) {
            const bf16_t* wRow = W + (size_t)(n0 + nc * 16 + l16) * K + k0 + half * 16;
            v16bf b = ld16(wRow);                       // B 32x16 fragment (contiguous)
            acc[nc] = wmma_bf16(a, b, acc[nc]);
        }
    }
#pragma unroll
    for (int nc = 0; nc < 4; ++nc) {
#pragma unroll
        for (int r = 0; r < 8; ++r) {
            size_t row = m0 + r + half * 8;
            size_t col = n0 + nc * 16 + l16;
            float v = acc[nc][r] * alpha;
            if (OUT_F32) ((float*)out)[row * N + col] = v;
            else         ((bf16_t*)out)[row * N + col] = f2bf(v);
        }
    }
}

// ---------------- kernel 4: flash attention with T5 bias ----------------
// Block = 128 threads (4 waves). Block owns 64 query rows of one (b,h);
// each wave owns 16 rows. kv tiles of 64 positions staged in LDS.
// Contraction index within a 32-chunk is permuted so each lane's pair of
// probabilities packs into one 32-bit LDS store; V is stored with the
// matching inverse permutation, so A(P) and B(V) orderings agree.
__global__ __launch_bounds__(128)
void attn_kernel(const bf16_t* __restrict__ Q, const bf16_t* __restrict__ Kg,
                 const bf16_t* __restrict__ Vg, const float* __restrict__ lut,
                 bf16_t* __restrict__ Ctx) {
    __shared__ __attribute__((aligned(16))) float  lut_s[LUT_LEN];
    __shared__ __attribute__((aligned(16))) bf16_t Ks[64 * 64];    // [kv][dk]
    __shared__ __attribute__((aligned(16))) bf16_t Vt[64 * 64];    // [dk][c] (transposed+permuted)
    __shared__ __attribute__((aligned(16))) bf16_t Ps[4 * 16 * 64]; // per-wave P tile [row][c]

    const int qBlocks = Ss / 64;
    const int qb = blockIdx.x % qBlocks;
    const int h  = (blockIdx.x / qBlocks) % Hh;
    const int b  = blockIdx.x / (qBlocks * Hh);
    const int tid  = threadIdx.x;
    const int lane = tid & 31;
    const int wave = tid >> 5;
    const int half = lane >> 4;
    const int l16  = lane & 15;

    // head-private bias LUT into LDS (visible after first loop barrier)
    const float* lutg = lut + h * LUT_LEN;
    for (int i = tid; i < LUT_LEN; i += 128) lut_s[i] = lutg[i];

    const int q0 = qb * 64 + wave * 16;
    const bf16_t* qptr = Q + (size_t)(b * Ss + q0 + l16) * Dd + h * DKk + half * 16;
    v16bf aQ0 = ld16(qptr);       // dk 0..31
    v16bf aQ1 = ld16(qptr + 32);       // dk 32..63

    v16bf ones;
#pragma unroll
    for (int i = 0; i < 16; ++i) ones[i] = f2bf(1.0f);

    float mrow[8];
    v8f o[4] = {};
    v8f lacc = {};
#pragma unroll
    for (int r = 0; r < 8; ++r) mrow[r] = -__builtin_inff();

    bf16_t* ps = Ps + wave * (16 * 64);

    // cooperative-load personalization: thread -> (kv row, 32-elem segment)
    const int cr   = tid >> 1;          // 0..63
    const int cseg = (tid & 1) * 32;    // 0 or 32
    // permuted contraction index for this physical kv row
    const int clocal = cr & 31;
    const int cperm  = (cr & 32) + ((clocal < 16) ? (2 * clocal) : (2 * (clocal - 16) + 1));

    for (int kt = 0; kt < Ss / 64; ++kt) {
        const int kv0 = kt * 64;
        __syncthreads();   // previous iteration's Ks/Vt readers are done
        {   // cooperative load: 64 kv rows x 64 dk
            const bf16_t* ksrc = Kg + (size_t)(b * Ss + kv0 + cr) * Dd + h * DKk + cseg;
            const bf16_t* vsrc = Vg + (size_t)(b * Ss + kv0 + cr) * Dd + h * DKk + cseg;
#pragma unroll
            for (int j = 0; j < 4; ++j)
                *(v8bf*)(Ks + cr * 64 + cseg + j * 8) = *(const v8bf*)(ksrc + j * 8);
            v8bf vv[4];
#pragma unroll
            for (int j = 0; j < 4; ++j) vv[j] = *(const v8bf*)(vsrc + j * 8);
#pragma unroll
            for (int i = 0; i < 32; ++i)
                Vt[(cseg + i) * 64 + cperm] = vv[i >> 3][i & 7];
        }
        __syncthreads();

        // scores: four 16x16 f32 tiles (kv columns t*16 .. t*16+15)
        v8f s[4];
#pragma unroll
        for (int t = 0; t < 4; ++t) {
            const bf16_t* kb = Ks + (size_t)(t * 16 + l16) * 64 + half * 16;
            v8f acc = {};
            acc = wmma_bf16(aQ0, ld16(kb),      acc);
            acc = wmma_bf16(aQ1, ld16(kb + 32), acc);
            s[t] = acc;
        }

        // bias + online softmax (rows r and r+8 live in the two half-waves)
#pragma unroll
        for (int r = 0; r < 8; ++r) {
            const int qm  = q0 + r + half * 8;
            const int idx = kv0 + l16 - qm + (Ss - 1);
            float x0 = s[0][r] + lut_s[idx];
            float x1 = s[1][r] + lut_s[idx + 16];
            float x2 = s[2][r] + lut_s[idx + 32];
            float x3 = s[3][r] + lut_s[idx + 48];
            float tm = rmax16(fmaxf(fmaxf(x0, x1), fmaxf(x2, x3)));
            float mn = fmaxf(mrow[r], tm);
            float corr = exp2f(mrow[r] - mn);
            mrow[r] = mn;
            float p0 = exp2f(x0 - mn);
            float p1 = exp2f(x1 - mn);
            float p2 = exp2f(x2 - mn);
            float p3 = exp2f(x3 - mn);
            lacc[r] *= corr;
#pragma unroll
            for (int nc = 0; nc < 4; ++nc) o[nc][r] *= corr;
            const int prow = r + half * 8;
            v2bf pa; pa[0] = f2bf(p0); pa[1] = f2bf(p1);   // logical c = 2*l16, 2*l16+1
            v2bf pb; pb[0] = f2bf(p2); pb[1] = f2bf(p3);   // logical c = 32+2*l16, +1
            *(v2bf*)(ps + prow * 64 + 2 * l16)      = pa;
            *(v2bf*)(ps + prow * 64 + 32 + 2 * l16) = pb;
        }
        // within-wave DS RAW hazard: stores above -> loads below
        asm volatile("s_wait_dscnt 0" ::: "memory");

        v16bf aP0 = ld16(ps + l16 * 64 + half * 16);
        v16bf aP1 = ld16(ps + l16 * 64 + 32 + half * 16);

        // row-sum of P via WMMA with a ones B-matrix (every column holds the sum)
        lacc = wmma_bf16(aP0, ones, lacc);
        lacc = wmma_bf16(aP1, ones, lacc);

#pragma unroll
        for (int nc = 0; nc < 4; ++nc) {
            const bf16_t* vb = Vt + (size_t)(nc * 16 + l16) * 64 + half * 16;
            o[nc] = wmma_bf16(aP0, ld16(vb),      o[nc]);   // O += P @ V (c chunk 0)
            o[nc] = wmma_bf16(aP1, ld16(vb + 32), o[nc]);   // O += P @ V (c chunk 1)
        }
    }

    // normalize and write context (bf16, feeds final GEMM)
#pragma unroll
    for (int r = 0; r < 8; ++r) {
        const float inv = 1.0f / lacc[r];
        bf16_t* cp = Ctx + (size_t)(b * Ss + q0 + r + half * 8) * Dd + h * DKk;
#pragma unroll
        for (int nc = 0; nc < 4; ++nc)
            cp[nc * 16 + l16] = f2bf(o[nc][r] * inv);
    }
}

// ---------------- launcher ----------------
extern "C" void kernel_launch(void* const* d_in, const int* in_sizes, int n_in,
                              void* d_out, int out_size, void* d_ws, size_t ws_size,
                              hipStream_t stream) {
    (void)in_sizes; (void)n_in; (void)out_size; (void)ws_size;
    const float* X   = (const float*)d_in[0];
    const float* Wq  = (const float*)d_in[1];
    const float* Wk  = (const float*)d_in[2];
    const float* Wv  = (const float*)d_in[3];
    const float* Wo  = (const float*)d_in[4];
    const float* rel = (const float*)d_in[5];
    float* out = (float*)d_out;

    const size_t nX = (size_t)Mm * Dd;   // 4,194,304
    const size_t nW = (size_t)Dd * Dd;   // 1,048,576

    bf16_t* ws  = (bf16_t*)d_ws;
    bf16_t* Xb  = ws;                 ws += nX;
    bf16_t* Wqb = ws;                 ws += nW;
    bf16_t* Wkb = ws;                 ws += nW;
    bf16_t* Wvb = ws;                 ws += nW;
    bf16_t* Wob = ws;                 ws += nW;
    bf16_t* Qb  = ws;                 ws += nX;
    bf16_t* Kb  = ws;                 ws += nX;
    bf16_t* Vb  = ws;                 ws += nX;
    bf16_t* Ctx = ws;                 ws += nX;
    float*  lut = (float*)ws;         // Hh * LUT_LEN floats

    // fp32 -> bf16 conversions
    cvt_f32_bf16<<<(int)((nX + 255) / 256), 256, 0, stream>>>(X,  Xb,  (int)nX);
    cvt_f32_bf16<<<(int)((nW + 255) / 256), 256, 0, stream>>>(Wq, Wqb, (int)nW);
    cvt_f32_bf16<<<(int)((nW + 255) / 256), 256, 0, stream>>>(Wk, Wkb, (int)nW);
    cvt_f32_bf16<<<(int)((nW + 255) / 256), 256, 0, stream>>>(Wv, Wvb, (int)nW);
    cvt_f32_bf16<<<(int)((nW + 255) / 256), 256, 0, stream>>>(Wo, Wob, (int)nW);

    // relative-position bias LUT (pre-scaled by log2e)
    bias_lut_kernel<<<(Hh * LUT_LEN + 255) / 256, 256, 0, stream>>>(rel, lut);

    // projections: (M/16)*(N/64) waves / 4 waves-per-block = 1024 blocks
    const int linBlocks = (Mm / 16) * (Dd / 64) / 4;
    // Q gets 1/sqrt(DK) AND log2e folded in (base-2 softmax)
    linear_wmma<false><<<linBlocks, 128, 0, stream>>>(Xb, Wqb, Qb, 0.125f * LOG2E, Mm, Dd, Dd);
    linear_wmma<false><<<linBlocks, 128, 0, stream>>>(Xb, Wkb, Kb, 1.0f,           Mm, Dd, Dd);
    linear_wmma<false><<<linBlocks, 128, 0, stream>>>(Xb, Wvb, Vb, 1.0f,           Mm, Dd, Dd);

    // attention: B*H*(S/64) = 1024 blocks of 4 waves
    attn_kernel<<<Bb * Hh * (Ss / 64), 128, 0, stream>>>(Qb, Kb, Vb, lut, Ctx);

    // output projection, f32 result
    linear_wmma<true><<<linBlocks, 128, 0, stream>>>(Ctx, Wob, out, 1.0f, Mm, Dd, Dd);
}
